// AttentionHead_1219770712599
// MI455X (gfx1250) — matrix operand order repaired and checked by reading, not compiled, not run on hardware
//
#include <hip/hip_runtime.h>

// ---------------------------------------------------------------------------
// AttentionHead for MI455X (gfx1250): fused f16-WMMA QKV projection +
// flash-attention. B=8, T=2048, C=1024, H=64. wave32.
// v3: 4-wave workgroups with double-buffered LDS staging of shared operands
// (W slabs in qkv, K/V tiles in attn) -> 4x less L2 traffic, and WMMA
// fragments are fed from LDS instead of stalling on s_wait_loadcnt 0.
// ---------------------------------------------------------------------------

typedef __attribute__((ext_vector_type(16))) _Float16 v16h;
typedef __attribute__((ext_vector_type(8)))  float    v8f;

#define LOG2E 1.4426950408889634f

static __device__ inline v8f v8f_zero() {
  v8f v;
#pragma unroll
  for (int i = 0; i < 8; ++i) v[i] = 0.0f;
  return v;
}

static __device__ inline v8f wmma16(v16h a, v16h b, v8f c) {
  return __builtin_amdgcn_wmma_f32_16x16x32_f16(
      /*neg_a=*/false, a, /*neg_b=*/false, b,
      /*c_mod=*/(short)0, c, /*reuse_a=*/false, /*reuse_b=*/false);
}

// 16 contiguous f16 -> B-fragment lane segment (K = (l/16)*16 .. +15)
static __device__ inline v16h ld16h(const _Float16* p) {
  union { int4 i[2]; v16h v; } u;
  u.i[0] = *reinterpret_cast<const int4*>(p);
  u.i[1] = *reinterpret_cast<const int4*>(p + 8);
  return u.v;
}

// A-fragment lane segment: 16 contiguous f16 at p (K = (l/16)*16 .. +15)
static __device__ inline v16h ldA(const _Float16* p) {
  union { int4 i[2]; v16h v; } u;
  u.i[0] = *reinterpret_cast<const int4*>(p);
  u.i[1] = *reinterpret_cast<const int4*>(p + 8);
  return u.v;
}

// ---------------------------------------------------------------------------
// Kernel 0: transpose+convert Wq/Wk/Wv  [C,H] fp32 -> Wt [3][H][C] f16
// ---------------------------------------------------------------------------
__global__ void wprep(const float* __restrict__ Wq, const float* __restrict__ Wk,
                      const float* __restrict__ Wv, _Float16* __restrict__ Wt) {
  int i = blockIdx.x * 256 + threadIdx.x;
  if (i >= 3 * 64 * 1024) return;
  int m = i >> 16;
  int r = i & 65535;
  int h = r >> 10;
  int c = r & 1023;
  const float* W = (m == 0) ? Wq : ((m == 1) ? Wk : Wv);
  Wt[i] = (_Float16)W[c * 64 + h];
}

// ---------------------------------------------------------------------------
// Kernel 1: fused QKV projection. Block = 4 waves = 64 rows; the 12KB W slab
// per 32-wide K-step is staged once per block into double-buffered LDS.
// Q,K stored [B*T, H] f16 (q pre-scaled by C^-0.5); V stored transposed
// [B, H, T] f16 so PV B-fragments are contiguous.
// ---------------------------------------------------------------------------
__global__ __launch_bounds__(128) void qkv_proj(const float* __restrict__ x,
                                                const _Float16* __restrict__ Wt,
                                                _Float16* __restrict__ Q,
                                                _Float16* __restrict__ Kmat,
                                                _Float16* __restrict__ Vt) {
  constexpr int C = 1024, H = 64, T = 2048;
  constexpr int SLAB = 3 * 64 * 32;                 // f16 elems per K-step slab
  __shared__ _Float16 Wst[2][SLAB];                 // 2 x 12KB

  const int tid  = threadIdx.x;                     // 0..127
  const int wv   = tid >> 5;                        // wave id 0..3
  const int lane = tid & 31;
  const int half = lane >> 4, lm = lane & 15;

  const int tile = blockIdx.x * 4 + wv;             // 16-row tile over B*T
  const float* xrow = x + (size_t)(tile * 16 + lm) * C;

  // cooperative stage of slab for K-step c0 into buffer p
  auto stage = [&](int c0, int p) {
    // 768 int4 slots: slot = (m*64+h)*4 + part, part covers 8 f16
#pragma unroll
    for (int s = 0; s < 6; ++s) {
      const int slot = tid + s * 128;
      const int m    = slot >> 8;
      const int rem  = slot & 255;
      const int h    = rem >> 2;
      const int part = rem & 3;
      const int4 v = *reinterpret_cast<const int4*>(
          Wt + (size_t)m * 65536 + (size_t)h * 1024 + c0 + part * 8);
      *reinterpret_cast<int4*>(&Wst[p][(m * 64 + h) * 32 + part * 8]) = v;
    }
  };

  v8f acc[3][4];
#pragma unroll
  for (int m = 0; m < 3; ++m)
#pragma unroll
    for (int n = 0; n < 4; ++n) acc[m][n] = v8f_zero();

  stage(0, 0);
  __syncthreads();

  for (int c0 = 0; c0 < C; c0 += 32) {
    const int p = (c0 >> 5) & 1;
    if (c0 + 32 < C) stage(c0 + 32, p ^ 1);

    // A-fragment from x (global, fp32 -> f16)
    const float* xp = xrow + c0 + half * 16;
    float4 xa = *reinterpret_cast<const float4*>(xp);
    float4 xb = *reinterpret_cast<const float4*>(xp + 4);
    float4 xc = *reinterpret_cast<const float4*>(xp + 8);
    float4 xd = *reinterpret_cast<const float4*>(xp + 12);
    v16h a;
    a[0]  = (_Float16)xa.x; a[1]  = (_Float16)xa.y; a[2]  = (_Float16)xa.z; a[3]  = (_Float16)xa.w;
    a[4]  = (_Float16)xb.x; a[5]  = (_Float16)xb.y; a[6]  = (_Float16)xb.z; a[7]  = (_Float16)xb.w;
    a[8]  = (_Float16)xc.x; a[9]  = (_Float16)xc.y; a[10] = (_Float16)xc.z; a[11] = (_Float16)xc.w;
    a[12] = (_Float16)xd.x; a[13] = (_Float16)xd.y; a[14] = (_Float16)xd.z; a[15] = (_Float16)xd.w;

    // B-fragments from LDS + WMMAs
#pragma unroll
    for (int m = 0; m < 3; ++m)
#pragma unroll
      for (int n = 0; n < 4; ++n) {
        v16h b = ld16h(&Wst[p][(m * 64 + n * 16 + lm) * 32 + half * 16]);
        acc[m][n] = wmma16(a, b, acc[m][n]);
      }
    __syncthreads();
  }

  const int row0 = tile * 16;
  const int bb = row0 / T;
  const int t0 = row0 % T;
#pragma unroll
  for (int n = 0; n < 4; ++n) {
    const int h = n * 16 + lm;
#pragma unroll
    for (int j = 0; j < 8; ++j) {
      const int row = row0 + j + half * 8;            // C-layout row
      Q[(size_t)row * H + h]    = (_Float16)(acc[0][n][j] * 0.03125f); // C^-0.5
      Kmat[(size_t)row * H + h] = (_Float16)acc[1][n][j];
    }
    union { _Float16 hh[8]; int4 i; } u;
#pragma unroll
    for (int j = 0; j < 8; ++j) u.hh[j] = (_Float16)acc[2][n][j];
    *reinterpret_cast<int4*>(Vt + ((size_t)(bb * H + h)) * T + t0 + half * 8) = u.i;
  }
}

// ---------------------------------------------------------------------------
// Kernel 2: flash attention. Block = 4 waves covering 64 consecutive queries
// (wave w owns queries Q0+w*16..+15). Key chunks of 32: K (32x64) and V
// (64x32) tiles are staged once per block into double-buffered LDS; each wave
// runs QK WMMAs -> online softmax -> P relayout (private LDS) -> PV WMMAs.
// Waves past their causal range skip compute but keep staging + barriers.
// ---------------------------------------------------------------------------
__global__ __launch_bounds__(128) void attn(const _Float16* __restrict__ Q,
                                            const _Float16* __restrict__ Kmat,
                                            const _Float16* __restrict__ Vt,
                                            float* __restrict__ out) {
  constexpr int H = 64, T = 2048;
  __shared__ _Float16 Kst[2][32 * 64];   // 2 x 4KB  [key][h]
  __shared__ _Float16 Vst[2][64 * 32];   // 2 x 4KB  [h][key]
  __shared__ _Float16 Pst[4][16 * 32];   // per-wave P tile [q][key]

  const int tid  = threadIdx.x;
  const int wv   = tid >> 5;
  const int lane = tid & 31;
  const int half = lane >> 4, lm = lane & 15;

  const int b  = blockIdx.x >> 5;        // 32 blocks of 64 queries per batch
  const int Q0 = (blockIdx.x & 31) * 64;
  const int q0 = Q0 + wv * 16;           // this wave's query tile

  const _Float16* Kb = Kmat + (size_t)b * T * H;
  const _Float16* Vb = Vt + (size_t)b * H * T;

  auto stage = [&](int kb, int p) {
    // K tile: 32 rows x 64 f16 = 256 int4 ; V tile: 64 rows x 32 f16 = 256 int4
#pragma unroll
    for (int s = 0; s < 2; ++s) {
      const int slot = tid + s * 128;          // 0..255
      const int row  = slot >> 3;              // key row
      const int part = slot & 7;
      *reinterpret_cast<int4*>(&Kst[p][row * 64 + part * 8]) =
          *reinterpret_cast<const int4*>(Kb + (size_t)(kb + row) * H + part * 8);
    }
#pragma unroll
    for (int s = 0; s < 2; ++s) {
      const int slot = tid + s * 128;          // 0..255
      const int h    = slot >> 2;
      const int part = slot & 3;
      *reinterpret_cast<int4*>(&Vst[p][h * 32 + part * 8]) =
          *reinterpret_cast<const int4*>(Vb + (size_t)h * T + kb + part * 8);
    }
  };

  const _Float16* qp = Q + ((size_t)(b * T + q0 + lm)) * H + half * 16;
  const v16h qa0 = ldA(qp);          // head dims 0..31
  const v16h qa1 = ldA(qp + 32);     // head dims 32..63

  v8f o[4];
  float m8[8], l8[8];
#pragma unroll
  for (int n = 0; n < 4; ++n) o[n] = v8f_zero();
#pragma unroll
  for (int j = 0; j < 8; ++j) { m8[j] = -1e30f; l8[j] = 0.0f; }

  const int nch_w   = (q0 + 15) / 32 + 1;      // chunks this wave needs
  const int nch_blk = (Q0 + 63) / 32 + 1;      // chunks the block stages

  stage(0, 0);
  __syncthreads();

  for (int kc = 0; kc < nch_blk; ++kc) {
    const int kb = kc * 32;
    const int p  = kc & 1;
    if (kc + 1 < nch_blk) stage((kc + 1) * 32, p ^ 1);

    if (kc < nch_w) {                          // wave-uniform guard
      // ---- QK^T from LDS ----
      v8f s0 = v8f_zero(), s1 = v8f_zero();
      s0 = wmma16(qa0, ld16h(&Kst[p][(lm) * 64 + half * 16]), s0);
      s0 = wmma16(qa1, ld16h(&Kst[p][(lm) * 64 + 32 + half * 16]), s0);
      s1 = wmma16(qa0, ld16h(&Kst[p][(16 + lm) * 64 + half * 16]), s1);
      s1 = wmma16(qa1, ld16h(&Kst[p][(16 + lm) * 64 + 32 + half * 16]), s1);

      if (kb + 31 > q0) {                      // causal mask (diagonal chunks)
#pragma unroll
        for (int j = 0; j < 8; ++j) {
          const int qrow = q0 + j + half * 8;
          if (kb + lm > qrow)      s0[j] = -1e30f;
          if (kb + 16 + lm > qrow) s1[j] = -1e30f;
        }
      }
      // row max across the 16 lanes holding each row
      float mn[8];
#pragma unroll
      for (int j = 0; j < 8; ++j) mn[j] = fmaxf(s0[j], s1[j]);
#pragma unroll
      for (int d = 8; d; d >>= 1)
#pragma unroll
        for (int j = 0; j < 8; ++j) mn[j] = fmaxf(mn[j], __shfl_xor(mn[j], d, 32));

      float al[8], rs[8];
      v8f p0, p1;
#pragma unroll
      for (int j = 0; j < 8; ++j) {
        const float mu = fmaxf(m8[j], mn[j]);
        al[j] = exp2f((m8[j] - mu) * LOG2E);
        m8[j] = mu;
        p0[j] = exp2f((s0[j] - mu) * LOG2E);
        p1[j] = exp2f((s1[j] - mu) * LOG2E);
        rs[j] = p0[j] + p1[j];
      }
#pragma unroll
      for (int d = 8; d; d >>= 1)
#pragma unroll
        for (int j = 0; j < 8; ++j) rs[j] += __shfl_xor(rs[j], d, 32);
#pragma unroll
      for (int j = 0; j < 8; ++j) l8[j] = l8[j] * al[j] + rs[j];
#pragma unroll
      for (int n = 0; n < 4; ++n)
#pragma unroll
        for (int j = 0; j < 8; ++j) o[n][j] *= al[j];

      // C-layout -> A-fragment relayout through private LDS region
#pragma unroll
      for (int j = 0; j < 8; ++j) {
        const int row = j + half * 8;
        Pst[wv][row * 32 + lm]      = (_Float16)p0[j];
        Pst[wv][row * 32 + 16 + lm] = (_Float16)p1[j];
      }
      asm volatile("s_wait_dscnt 0" ::: "memory");   // RAW within wave
      const v16h pa = ldA(&Pst[wv][lm * 32 + half * 16]);

      // ---- PV from LDS ----
#pragma unroll
      for (int n = 0; n < 4; ++n)
        o[n] = wmma16(pa, ld16h(&Vst[p][(n * 16 + lm) * 32 + half * 16]), o[n]);
    }
    __syncthreads();
  }

  float inv[8];
#pragma unroll
  for (int j = 0; j < 8; ++j) inv[j] = 1.0f / l8[j];
  float* op = out + ((size_t)(b * T + q0)) * H + lm;
#pragma unroll
  for (int n = 0; n < 4; ++n)
#pragma unroll
    for (int j = 0; j < 8; ++j)
      op[(size_t)(j + half * 8) * H + n * 16] = o[n][j] * inv[j];
}

// ---------------------------------------------------------------------------
extern "C" void kernel_launch(void* const* d_in, const int* in_sizes, int n_in,
                              void* d_out, int out_size, void* d_ws, size_t ws_size,
                              hipStream_t stream) {
  (void)in_sizes; (void)n_in; (void)out_size; (void)ws_size;
  const float* x  = (const float*)d_in[0];
  const float* Wq = (const float*)d_in[1];
  const float* Wk = (const float*)d_in[2];
  const float* Wv = (const float*)d_in[3];
  float* out = (float*)d_out;

  // workspace carve (f16): Wt 3*64*1024, Q/K/Vt each 8*2048*64 -> ~6.4 MB
  _Float16* Wt = (_Float16*)d_ws;
  _Float16* Q  = Wt + 3 * 64 * 1024;
  _Float16* K  = Q + 8 * 2048 * 64;
  _Float16* Vt = K + 8 * 2048 * 64;

  wprep<<<(3 * 64 * 1024 + 255) / 256, 256, 0, stream>>>(Wq, Wk, Wv, Wt);
  qkv_proj<<<(8 * 2048) / 64, 128, 0, stream>>>(x, Wt, Q, K, Vt);
  attn<<<8 * (2048 / 64), 128, 0, stream>>>(Q, K, Vt, out);
}
